// GAT_Model_42279658061893
// MI455X (gfx1250) — matrix operand order repaired and checked by reading, not compiled, not run on hardware
//
#include <hip/hip_runtime.h>
#include <hip/hip_bf16.h>

#define NEG_SLOPE 0.2f

typedef __attribute__((ext_vector_type(16))) __bf16 v16bf;
typedef __attribute__((ext_vector_type(8)))  float  v8f;

union BF16Frag { v16bf v; unsigned u[8]; };

static __device__ __forceinline__ unsigned f32_to_bf16u(float f) {
  unsigned u = __float_as_uint(f);
  unsigned r = 0x7FFFu + ((u >> 16) & 1u);
  return (u + r) >> 16;
}
static __device__ __forceinline__ unsigned short f32_to_bf16(float f) {
  return (unsigned short)f32_to_bf16u(f);
}
static __device__ __forceinline__ unsigned mapf(float x) {
  unsigned u = __float_as_uint(x);
  return (u & 0x80000000u) ? ~u : (u | 0x80000000u);
}
static __device__ __forceinline__ float unmapf(unsigned u) {
  unsigned b = (u & 0x80000000u) ? (u ^ 0x80000000u) : ~u;
  return __uint_as_float(b);
}
static __device__ __forceinline__ bool finitef(float x) {
  return ((__float_as_uint(x) >> 23) & 0xFFu) != 0xFFu;
}

// ---------------------------------------------------------------------------
// Y[N,M] = X[N,K] @ W[K,M] + bias[M]   (bf16 WMMA, f32 accumulate)
// block = 256 threads = 8 waves; block tile = 128 rows x 32 cols;
// each wave: one A fragment, two B fragments -> 2x v_wmma per K-step.
// K multiple of 32, M multiple of 32, col tiles fully covered by grid.y.
// ---------------------------------------------------------------------------
__global__ __launch_bounds__(256) void gemm_bias_bf16_wmma(
    const float* __restrict__ X, const float* __restrict__ W,
    const float* __restrict__ bias, float* __restrict__ Y,
    int N, int K, int M)
{
  __shared__ unsigned short Xs[128][32];   // [row][k]  bf16, 8KB
  __shared__ unsigned short Wt[32][32];    // [n][k]    bf16 (transposed), 2KB

  const int tid  = threadIdx.x;
  const int lane = tid & 31;
  const int wave = tid >> 5;               // 0..7
  const int row0 = blockIdx.x * 128;
  const int col0 = blockIdx.y * 32;

  v8f acc0 = {}, acc1 = {};

  for (int kk = 0; kk < K; kk += 32) {
    // ---- X tile: 128x32 f32 -> bf16, branchless (clamped rows), float4 loads
    #pragma unroll
    for (int i = 0; i < 4; ++i) {
      int idx4 = i * 256 + tid;            // 0..1023 float4 slots
      int r = idx4 >> 3, c4 = idx4 & 7;
      int gr = row0 + r;
      gr = (gr < N) ? gr : (N - 1);        // clamp: garbage rows are store-guarded
      const float* src = X + (long)gr * K + kk + c4 * 4;
      if (kk + 32 < K) __builtin_prefetch(src + 32, 0, 1);
      float4 v = *(const float4*)src;
      uint2 p;
      p.x = f32_to_bf16u(v.x) | (f32_to_bf16u(v.y) << 16);
      p.y = f32_to_bf16u(v.z) | (f32_to_bf16u(v.w) << 16);
      *(uint2*)&Xs[r][c4 * 4] = p;
    }
    // ---- W tile: 32x32, transposed into Wt[n][k]; one float4 per thread
    {
      int k = tid >> 3, n4 = tid & 7;
      float4 v = *(const float4*)(W + (long)(kk + k) * M + col0 + n4 * 4);
      Wt[n4 * 4 + 0][k] = f32_to_bf16(v.x);
      Wt[n4 * 4 + 1][k] = f32_to_bf16(v.y);
      Wt[n4 * 4 + 2][k] = f32_to_bf16(v.z);
      Wt[n4 * 4 + 3][k] = f32_to_bf16(v.w);
    }
    __syncthreads();

    const int half = lane >> 4;
    const int mrow = (lane & 15) + wave * 16;

    // A fragment: 16-bit A 16x32 layout (ISA 7.12.2): VGPR p holds packed
    // K pair {2p,2p+1} (+8 for lanes 16-31; +16 for p>=4)
    BF16Frag af, bf0, bf1;
    #pragma unroll
    for (int p = 0; p < 8; ++p) {
      int k0 = (p < 4) ? (2 * p + 8 * half) : (16 + 2 * (p - 4) + 8 * half);
      af.u[p] = *(const unsigned*)&Xs[mrow][k0];
    }
    // B fragments: same K order as A (pair p: K=2p+8*half, +16 for p>=4), col=lane%16
    const unsigned* w0 = (const unsigned*)&Wt[lane & 15][8 * half];
    const unsigned* w1 = (const unsigned*)&Wt[16 + (lane & 15)][8 * half];
    #pragma unroll
    for (int p = 0; p < 8; ++p) { int q = (p < 4) ? p : (p + 4); bf0.u[p] = w0[q]; bf1.u[p] = w1[q]; }

    acc0 = __builtin_amdgcn_wmma_f32_16x16x32_bf16(
        false, af.v, false, bf0.v, (short)0, acc0, false, false);
    acc1 = __builtin_amdgcn_wmma_f32_16x16x32_bf16(
        false, af.v, false, bf1.v, (short)0, acc1, false, false);
    __syncthreads();
  }

  // C/D layout: VGPR i -> M = i (lanes 0-15) / i+8 (lanes 16-31); N = lane%16
  const int half = lane >> 4;
  const int c0 = col0 + (lane & 15);
  const int c1 = c0 + 16;
  const float bv0 = bias[c0];
  const float bv1 = bias[c1];
  #pragma unroll
  for (int i = 0; i < 8; ++i) {
    int row = row0 + wave * 16 + i + 8 * half;
    if (row < N) {
      Y[(long)row * M + c0] = acc0[i] + bv0;
      Y[(long)row * M + c1] = acc1[i] + bv1;
    }
  }
}

// ---------------------------------------------------------------------------
// generic 32-bit fill
// ---------------------------------------------------------------------------
__global__ void fill_u32(unsigned* __restrict__ p, unsigned v, long n) {
  long t = (long)blockIdx.x * blockDim.x + threadIdx.x;
  if (t < n) p[t] = v;
}

// ---------------------------------------------------------------------------
// per-edge attention logits + segment max (int-bits atomicMax)
// ---------------------------------------------------------------------------
__global__ void edge_logits_max(
    const float* __restrict__ xl, const float* __restrict__ xr,
    const float* __restrict__ att, const int* __restrict__ ei,
    int E, int Etot, int H, int C,
    float* __restrict__ logits, unsigned* __restrict__ mbits)
{
  long e = (long)blockIdx.x * blockDim.x + threadIdx.x;
  if (e >= Etot) return;
  int s, d; bool valid;
  if (e < E) { s = ei[e]; d = ei[(long)E + e]; valid = (s != d); }
  else       { s = d = (int)(e - E); valid = true; }

  const int HC = H * C;
  const float4* pl = (const float4*)(xl + (long)s * HC);
  const float4* pr = (const float4*)(xr + (long)d * HC);

  for (int h = 0; h < H; ++h) {
    float lg = -__builtin_inff();
    if (valid) {
      const float4* pa = (const float4*)(att + h * C);
      float acc = 0.f;
      for (int c4 = 0; c4 < C / 4; ++c4) {
        float4 a = pl[h * (C / 4) + c4];
        float4 r = pr[h * (C / 4) + c4];
        float4 w = pa[c4];
        float t;
        t = a.x + r.x; acc += (t > 0.f ? t : NEG_SLOPE * t) * w.x;
        t = a.y + r.y; acc += (t > 0.f ? t : NEG_SLOPE * t) * w.y;
        t = a.z + r.z; acc += (t > 0.f ? t : NEG_SLOPE * t) * w.z;
        t = a.w + r.w; acc += (t > 0.f ? t : NEG_SLOPE * t) * w.w;
      }
      lg = acc;
      atomicMax(&mbits[(long)d * H + h], mapf(lg));
    }
    logits[e * H + h] = lg;
  }
}

// ---------------------------------------------------------------------------
// exp(logit - max) and denominator accumulation
// ---------------------------------------------------------------------------
__global__ void edge_exp_sum(
    const float* __restrict__ logits, const unsigned* __restrict__ mbits,
    const int* __restrict__ ei, int E, int Etot, int H,
    float* __restrict__ ex, float* __restrict__ denom)
{
  long e = (long)blockIdx.x * blockDim.x + threadIdx.x;
  if (e >= Etot) return;
  int d = (e < E) ? ei[(long)E + e] : (int)(e - E);
  for (int h = 0; h < H; ++h) {
    float m = unmapf(mbits[(long)d * H + h]);
    if (!finitef(m)) m = 0.f;
    float l = logits[e * H + h];
    float v = finitef(l) ? __expf(l - m) : 0.f;
    ex[e * H + h] = v;
    if (v != 0.f) atomicAdd(&denom[(long)d * H + h], v);
  }
}

// ---------------------------------------------------------------------------
// numerator scatter: num[dst] += ex * xl[src]  (4 features per thread)
// ---------------------------------------------------------------------------
__global__ void edge_accum(
    const float* __restrict__ ex, const float* __restrict__ xl,
    const int* __restrict__ ei, int E, long total, int H, int C,
    float* __restrict__ num)
{
  long t = (long)blockIdx.x * blockDim.x + threadIdx.x;
  if (t >= total) return;
  const int q = (H * C) / 4;
  long e = t / q;
  int  f = (int)(t - e * q);
  int s, d;
  if (e < E) { s = ei[e]; d = ei[(long)E + e]; if (s == d) return; }
  else       { s = d = (int)(e - E); }
  int h = (f * 4) / C;
  float a = ex[e * H + h];
  if (a == 0.f) return;
  float4 xj = ((const float4*)(xl + (long)s * H * C))[f];
  float* o = num + (long)d * H * C + f * 4;
  atomicAdd(o + 0, a * xj.x);
  atomicAdd(o + 1, a * xj.y);
  atomicAdd(o + 2, a * xj.z);
  atomicAdd(o + 3, a * xj.w);
}

// ---------------------------------------------------------------------------
// node finalize: divide by denom, add bias, relu (concat or head-mean)
// ---------------------------------------------------------------------------
__global__ void finalize_concat(
    const float* __restrict__ num, const float* __restrict__ denom,
    const float* __restrict__ bias, float* __restrict__ out,
    int N, int H, int C)
{
  long t = (long)blockIdx.x * blockDim.x + threadIdx.x;
  long total = (long)N * H * C;
  if (t >= total) return;
  int HC = H * C;
  long n = t / HC;
  int  j = (int)(t - n * HC);
  int  h = j / C;
  float v = num[t] / denom[n * H + h] + bias[j];
  out[t] = v > 0.f ? v : 0.f;
}

__global__ void finalize_mean(
    const float* __restrict__ num, const float* __restrict__ denom,
    const float* __restrict__ bias, float* __restrict__ out,
    int N, int H, int C)
{
  long t = (long)blockIdx.x * blockDim.x + threadIdx.x;
  long total = (long)N * C;
  if (t >= total) return;
  long n = t / C;
  int  c = (int)(t - n * C);
  float s = 0.f;
  for (int h = 0; h < H; ++h)
    s += num[n * H * C + h * C + c] / denom[n * H + h];
  float v = s / (float)H + bias[c];
  out[n * C + c] = v > 0.f ? v : 0.f;
}

// ---------------------------------------------------------------------------
// graph pooling: per-graph max (bits), sum and count
// ---------------------------------------------------------------------------
__global__ void pool_graphs(
    const float* __restrict__ h, const int* __restrict__ batch,
    unsigned* __restrict__ gmx, float* __restrict__ gsum,
    float* __restrict__ cnt, int N)
{
  long t = (long)blockIdx.x * blockDim.x + threadIdx.x;
  if (t >= (long)N * 64) return;
  long n = t >> 6;
  int  j = (int)(t & 63);
  int  g = batch[n];
  float v = h[n * 64 + j];
  atomicMax(&gmx[g * 64 + j], mapf(v));
  atomicAdd(&gsum[g * 64 + j], v);
  if (j == 0) atomicAdd(&cnt[g], 1.0f);
}

// ---------------------------------------------------------------------------
// classifier head: out[64,10] = [gmx | gmean] @ Wout + bout
// ---------------------------------------------------------------------------
__global__ void head_linear(
    const unsigned* __restrict__ gmx, const float* __restrict__ gsum,
    const float* __restrict__ cnt, const float* __restrict__ Wout,
    const float* __restrict__ bout, float* __restrict__ out, int NC)
{
  int t = blockIdx.x * blockDim.x + threadIdx.x;
  if (t >= 64 * NC) return;
  int g = t / NC;
  int k = t - g * NC;
  float c = cnt[g];
  if (c < 1.f) c = 1.f;
  float acc = bout[k];
  for (int j = 0; j < 64; ++j) {
    float mx = unmapf(gmx[g * 64 + j]);
    if (!finitef(mx)) mx = 0.f;
    acc += mx * Wout[j * NC + k];
    acc += (gsum[g * 64 + j] / c) * Wout[(64 + j) * NC + k];
  }
  out[g * NC + k] = acc;
}

// ---------------------------------------------------------------------------
static inline unsigned cdivu(long a, long b) { return (unsigned)((a + b - 1) / b); }

extern "C" void kernel_launch(void* const* d_in, const int* in_sizes, int n_in,
                              void* d_out, int out_size, void* d_ws, size_t ws_size,
                              hipStream_t stream) {
  const float* x     = (const float*)d_in[0];
  const int*   ei    = (const int*)d_in[1];
  const int*   batch = (const int*)d_in[2];
  const int N    = in_sizes[2];
  const int E    = in_sizes[1] / 2;
  const int Etot = E + N;
  const int F    = in_sizes[0] / N;

  const float* Wl[3]   = {(const float*)d_in[3],  (const float*)d_in[9],  (const float*)d_in[15]};
  const float* blb[3]  = {(const float*)d_in[4],  (const float*)d_in[10], (const float*)d_in[16]};
  const float* Wr[3]   = {(const float*)d_in[5],  (const float*)d_in[11], (const float*)d_in[17]};
  const float* brb[3]  = {(const float*)d_in[6],  (const float*)d_in[12], (const float*)d_in[18]};
  const float* att[3]  = {(const float*)d_in[7],  (const float*)d_in[13], (const float*)d_in[19]};
  const float* bias[3] = {(const float*)d_in[8],  (const float*)d_in[14], (const float*)d_in[20]};
  const float* Wout = (const float*)d_in[21];
  const float* bout = (const float*)d_in[22];

  const int Kin[3] = {F, 192, 128};
  const int Hh[3]  = {3, 2, 2};
  const int Cc[3]  = {64, 64, 64};

  char* ws = (char*)d_ws;
  size_t off = 0;
  auto salloc = [&](size_t bytes) -> void* {
    void* p = ws + off;
    off += (bytes + 255) & ~(size_t)255;
    return p;
  };
  float*    xl     = (float*)   salloc((size_t)N * 192 * 4);
  float*    xr     = (float*)   salloc((size_t)N * 192 * 4);
  float*    num    = (float*)   salloc((size_t)N * 192 * 4);
  float*    hbuf   = (float*)   salloc((size_t)N * 192 * 4);
  float*    logits = (float*)   salloc((size_t)Etot * 3 * 4);
  float*    ex     = (float*)   salloc((size_t)Etot * 3 * 4);
  unsigned* mbits  = (unsigned*)salloc((size_t)N * 3 * 4);
  float*    denom  = (float*)   salloc((size_t)N * 3 * 4);
  unsigned* gmx    = (unsigned*)salloc((size_t)64 * 64 * 4);
  float*    gsum   = (float*)   salloc((size_t)64 * 64 * 4);
  float*    cnt    = (float*)   salloc((size_t)64 * 4);

  const float* hin = x;
  for (int l = 0; l < 3; ++l) {
    const int K = Kin[l], H = Hh[l], C = Cc[l], M = H * C;

    dim3 gg(cdivu(N, 128), M / 32);
    gemm_bias_bf16_wmma<<<gg, 256, 0, stream>>>(hin, Wl[l], blb[l], xl, N, K, M);
    gemm_bias_bf16_wmma<<<gg, 256, 0, stream>>>(hin, Wr[l], brb[l], xr, N, K, M);

    fill_u32<<<cdivu((long)N * H, 256), 256, 0, stream>>>(mbits, 0u, (long)N * H);
    fill_u32<<<cdivu((long)N * H, 256), 256, 0, stream>>>((unsigned*)denom, 0u, (long)N * H);
    fill_u32<<<cdivu((long)N * M, 256), 256, 0, stream>>>((unsigned*)num, 0u, (long)N * M);

    edge_logits_max<<<cdivu(Etot, 256), 256, 0, stream>>>(
        xl, xr, att[l], ei, E, Etot, H, C, logits, mbits);
    edge_exp_sum<<<cdivu(Etot, 256), 256, 0, stream>>>(
        logits, mbits, ei, E, Etot, H, ex, denom);

    long tot = (long)Etot * (M / 4);
    edge_accum<<<cdivu(tot, 256), 256, 0, stream>>>(ex, xl, ei, E, tot, H, C, num);

    if (l < 2)
      finalize_concat<<<cdivu((long)N * M, 256), 256, 0, stream>>>(
          num, denom, bias[l], hbuf, N, H, C);
    else
      finalize_mean<<<cdivu((long)N * C, 256), 256, 0, stream>>>(
          num, denom, bias[l], hbuf, N, H, C);
    hin = hbuf;
  }

  fill_u32<<<cdivu(64 * 64, 256), 256, 0, stream>>>(gmx, 0u, 64 * 64);
  fill_u32<<<cdivu(64 * 64, 256), 256, 0, stream>>>((unsigned*)gsum, 0u, 64 * 64);
  fill_u32<<<1, 64, 0, stream>>>((unsigned*)cnt, 0u, 64);

  pool_graphs<<<cdivu((long)N * 64, 256), 256, 0, stream>>>(hbuf, batch, gmx, gsum, cnt, N);

  const int NC = out_size / 64;   // 10
  head_linear<<<cdivu(64 * NC, 256), 256, 0, stream>>>(gmx, gsum, cnt, Wout, bout,
                                                       (float*)d_out, NC);
}
